// MultiHeadAttention_8546984919390
// MI455X (gfx1250) — matrix operand order repaired and checked by reading, not compiled, not run on hardware
//
#include <hip/hip_runtime.h>

// ---------------------------------------------------------------------------
// CDNA5 (gfx1250) wave32 WMMA bf16 attention pipeline.
// ---------------------------------------------------------------------------

typedef __bf16 bf16_t;
typedef __attribute__((ext_vector_type(16))) __bf16        v16bf;
typedef __attribute__((ext_vector_type(8)))  float         v8f;
typedef __attribute__((ext_vector_type(4)))  unsigned int  u32x4;
typedef __attribute__((ext_vector_type(4)))  float         f32x4;

union Frag16 { u32x4 u[2]; v16bf v; };

__device__ __forceinline__ v8f wmma_bf16(v16bf a, v16bf b, v8f c) {
  // D(16x16,f32) = A(16x32,bf16) * B(32x16,bf16) + C
  return __builtin_amdgcn_wmma_f32_16x16x32_bf16(
      /*neg_a=*/false, a, /*neg_b=*/false, b,
      /*c_mod=*/(short)0, c, /*reuse_a=*/false, /*reuse_b=*/false);
}

// Async global->LDS copy of 16 bytes per lane (ASYNCcnt-tracked, no VGPRs).
// LDS byte offset = low 32 bits of the generic pointer (ISA 10.2 aperture rule).
__device__ __forceinline__ void async_ld_b128(const bf16_t* gsrc, bf16_t* lds_dst) {
  unsigned loff = (unsigned)(unsigned long long)(uintptr_t)lds_dst;
  asm volatile("global_load_async_to_lds_b128 %0, %1, off"
               :: "v"(loff), "v"(gsrc)
               : "memory");
}
__device__ __forceinline__ void wait_asynccnt0() {
  asm volatile("s_wait_asynccnt 0x0" ::: "memory");
}

// A fragment: lane = {row = lane&15, khalf = lane>>4};
// elements 0..7  <-> K = khalf*8 + 0..7
// elements 8..15 <-> K = 16 + khalf*8 + 0..7
// LDS tile stored [row][k] with row stride `lda` (multiple of 8 -> 16B aligned)
__device__ __forceinline__ v16bf load_frag_a(const bf16_t* base, int lda) {
  const int lane = threadIdx.x & 31;
  const int row = lane & 15, kh = lane >> 4;
  Frag16 f;
  f.u[0] = *(const u32x4*)(base + row * lda + kh * 8);
  f.u[1] = *(const u32x4*)(base + row * lda + 16 + kh * 8);
  return f.v;
}

// B fragment: lane = {col = lane&15, khalf = lane>>4}; elements 0..7 <-> K = khalf*8 + 0..7, 8..15 <-> K = 16 + khalf*8 + 0..7
// LDS tile stored transposed [col][k] with row stride `ldb`
__device__ __forceinline__ v16bf load_frag_b(const bf16_t* base, int ldb) {
  const int lane = threadIdx.x & 31;
  const int col = lane & 15, kh = lane >> 4;
  Frag16 f;
  const bf16_t* p = base + col * ldb + kh * 8;
  f.u[0] = *(const u32x4*)(p);
  f.u[1] = *(const u32x4*)(p + 16);
  return f.v;
}

// ---------------------------------------------------------------------------
// Generic 2048x2048x2048 GEMM (per batch z), bf16 WMMA, f32 accumulate.
//   A_IS_F32: A read as f32 (x) or bf16 (attention output O)
//   OUT_MODE 0: C = bf16, scatter to [b][h=n>>7][row][d=n&127]  (q/k/v)
//   OUT_MODE 1: C = f32,  [b][row][n]                           (final)
// Block tile 128x128, K-step 32, double-buffered LDS (fill nxt || WMMA cur).
// 256 threads = 8 waves (4x2); each wave computes a 32x64 subtile = 2x4 accs.
// ---------------------------------------------------------------------------
template <bool A_IS_F32, int OUT_MODE>
__global__ __launch_bounds__(256) void gemm2048_kernel(const void* __restrict__ Av,
                                                       const float* __restrict__ B,
                                                       void* __restrict__ Cv) {
  constexpr int KT = 2048, NT = 2048;
  __shared__ bf16_t As[2][128 * 40];   // [m][k], pad 32->40 (conflict-free b128)
  __shared__ bf16_t Bs[2][128 * 40];   // [n][k] (transposed)

  const int tid = threadIdx.x;
  const int bz = blockIdx.z;
  const int m0 = blockIdx.y * 128;
  const int n0 = blockIdx.x * 128;
  const size_t abatch = (size_t)bz * 2048 * 2048;

  const int wave = tid >> 5;
  const int wr = wave >> 1;  // 0..3 -> 32-row group
  const int wc = wave & 1;   // 0..1 -> 64-col group
  const int lane = tid & 31;

  auto fillA = [&](int buf, int k0) {
    if constexpr (A_IS_F32) {
      const float* Af = (const float*)Av + abatch;
      for (int p = 0; p < 4; ++p) {
        int idx = p * 256 + tid;          // 0..1023
        int r = idx >> 3, kq = (idx & 7) * 4;
        f32x4 val = *(const f32x4*)(Af + (size_t)(m0 + r) * KT + k0 + kq);
        As[buf][r * 40 + kq + 0] = (bf16_t)val[0];
        As[buf][r * 40 + kq + 1] = (bf16_t)val[1];
        As[buf][r * 40 + kq + 2] = (bf16_t)val[2];
        As[buf][r * 40 + kq + 3] = (bf16_t)val[3];
      }
    } else {
      const bf16_t* Af = (const bf16_t*)Av + abatch;
      for (int p = 0; p < 2; ++p) {
        int idx = p * 256 + tid;          // 0..511
        int r = idx >> 2, c8 = (idx & 3) * 8;
        async_ld_b128(Af + (size_t)(m0 + r) * KT + k0 + c8, &As[buf][r * 40 + c8]);
      }
    }
  };
  auto fillB = [&](int buf, int k0) {
    // global [k][n] f32 -> LDS [n][k] bf16 (transpose + convert)
    for (int p = 0; p < 4; ++p) {
      int idx = p * 256 + tid;            // 0..1023
      int kk2 = idx >> 5, nq = (idx & 31) * 4;
      f32x4 val = *(const f32x4*)(B + (size_t)(k0 + kk2) * NT + n0 + nq);
      Bs[buf][(nq + 0) * 40 + kk2] = (bf16_t)val[0];
      Bs[buf][(nq + 1) * 40 + kk2] = (bf16_t)val[1];
      Bs[buf][(nq + 2) * 40 + kk2] = (bf16_t)val[2];
      Bs[buf][(nq + 3) * 40 + kk2] = (bf16_t)val[3];
    }
  };

  v8f acc[2][4];
  for (int r = 0; r < 2; ++r)
    for (int c = 0; c < 4; ++c)
      for (int i = 0; i < 8; ++i) acc[r][c][i] = 0.f;

  fillA(0, 0);
  fillB(0, 0);
  if constexpr (!A_IS_F32) wait_asynccnt0();
  __syncthreads();

#pragma unroll 1
  for (int k0 = 0; k0 < KT; k0 += 32) {
    const int cur = (k0 >> 5) & 1;
    if (k0 + 32 < KT) {                 // prefetch next K-slice into other buffer
      fillA(cur ^ 1, k0 + 32);
      fillB(cur ^ 1, k0 + 32);
    }
    v16bf af[2], bfr[4];
    for (int r = 0; r < 2; ++r) af[r] = load_frag_a(&As[cur][(wr * 32 + r * 16) * 40], 40);
    for (int c = 0; c < 4; ++c) bfr[c] = load_frag_b(&Bs[cur][(wc * 64 + c * 16) * 40], 40);
    for (int r = 0; r < 2; ++r)
      for (int c = 0; c < 4; ++c) acc[r][c] = wmma_bf16(af[r], bfr[c], acc[r][c]);
    if constexpr (!A_IS_F32) wait_asynccnt0();
    __syncthreads();
  }

  // C/D layout: lane holds col = lane&15; VGPR i holds row = i + 8*(lane>>4)
  const int col = lane & 15, kh = lane >> 4;
  for (int r = 0; r < 2; ++r)
    for (int c = 0; c < 4; ++c)
      for (int i = 0; i < 8; ++i) {
        int row = m0 + wr * 32 + r * 16 + i + 8 * kh;
        int n = n0 + wc * 64 + c * 16 + col;
        if constexpr (OUT_MODE == 0) {
          bf16_t* C = (bf16_t*)Cv;  // [B][H][T][D]
          C[(((size_t)bz * 16 + (n >> 7)) * 2048 + row) * 128 + (n & 127)] =
              (bf16_t)acc[r][c][i];
        } else {
          float* C = (float*)Cv;    // [B][T][M]
          C[((size_t)bz * 2048 + row) * 2048 + n] = acc[r][c][i];
        }
      }
}

// ---------------------------------------------------------------------------
// RoPE (in-place, q and k, bf16 [B][H][T][D], half = 64)
// ---------------------------------------------------------------------------
__global__ __launch_bounds__(64) void rope_kernel(bf16_t* __restrict__ q,
                                                  bf16_t* __restrict__ k) {
  const int d = threadIdx.x;  // 0..63
  const int t = blockIdx.x;
  const int h = blockIdx.y;
  const int b = blockIdx.z;
  const size_t base = (((size_t)(b * 16 + h)) * 2048 + t) * 128;
  // 10000^(-d/64) = exp(-d * ln(10000)/64)
  const float inv = __expf(-(float)d * 0.14391157f);
  float sn, cs;
  __sincosf((float)t * inv, &sn, &cs);
  float qe = (float)q[base + d], qo = (float)q[base + d + 64];
  q[base + d]      = (bf16_t)(qe * cs - qo * sn);
  q[base + d + 64] = (bf16_t)(qe * sn + qo * cs);
  float ke = (float)k[base + d], ko = (float)k[base + d + 64];
  k[base + d]      = (bf16_t)(ke * cs - ko * sn);
  k[base + d + 64] = (bf16_t)(ke * sn + ko * cs);
}

// ---------------------------------------------------------------------------
// Flash attention (causal), per (b,h). 128 threads = 4 waves; each wave owns
// 16 query rows; workgroup tiles 64 queries x 64 keys per iteration.
// K tile staged via async global->LDS copies (ASYNCcnt); V^T staged manually.
// Output O written bf16 [B][T][H*D] for the output projection GEMM.
// ---------------------------------------------------------------------------
__global__ __launch_bounds__(128) void attn_kernel(const bf16_t* __restrict__ q,
                                                   const bf16_t* __restrict__ kg,
                                                   const bf16_t* __restrict__ vg,
                                                   bf16_t* __restrict__ og) {
  constexpr int T = 2048, D = 128, H = 16;
  __shared__ bf16_t Ks[64 * 136];    // [j][d], pad 128->136
  __shared__ bf16_t Vs[128 * 72];    // [d][j] (transposed), pad 64->72
  __shared__ bf16_t Ps[4][16 * 72];  // per-wave P transpose buffer

  const int tid = threadIdx.x;
  const int wave = tid >> 5;
  const int lane = tid & 31;
  const int col = lane & 15, kh = lane >> 4;
  const int i0 = blockIdx.x * 64;
  const int h = blockIdx.y, b = blockIdx.z;
  const int i0w = i0 + wave * 16;  // this wave's first query row (uniform)

  const size_t headOff = ((size_t)(b * H + h)) * T * D;
  const bf16_t* qh = q + headOff;
  const bf16_t* khp = kg + headOff;
  const bf16_t* vhp = vg + headOff;

  // Q A-fragments for this wave's 16 rows, all 4 d-steps (held in VGPRs)
  v16bf qf[4];
  {
    const bf16_t* qrow = qh + (size_t)(i0w + col) * D;  // row = lane&15
    for (int s = 0; s < 4; ++s) {
      Frag16 f;
      f.u[0] = *(const u32x4*)(qrow + s * 32 + kh * 8);
      f.u[1] = *(const u32x4*)(qrow + s * 32 + 16 + kh * 8);
      qf[s] = f.v;
    }
  }

  float mrow[8], lrow[8];
  v8f oacc[8];
  for (int i = 0; i < 8; ++i) { mrow[i] = -1e30f; lrow[i] = 0.f; }
  for (int dt = 0; dt < 8; ++dt)
    for (int i = 0; i < 8; ++i) oacc[dt][i] = 0.f;

  const int nblocks = i0 / 64 + 1;  // causal: only key blocks with j0 <= i0
#pragma unroll 1
  for (int jb = 0; jb < nblocks; ++jb) {
    const int j0 = jb * 64;
    // ---- stage K tile [64][128]: straight bf16 copy -> async to LDS ----
    for (int p = 0; p < 8; ++p) {
      int idx = p * 128 + tid;  // 0..1023 chunks of 8 bf16
      int r = idx >> 4, c8 = (idx & 15) * 8;
      async_ld_b128(khp + (size_t)(j0 + r) * D + c8, &Ks[r * 136 + c8]);
    }
    // ---- stage V^T tile [128][64] (transpose on fill) ----
    for (int p = 0; p < 8; ++p) {
      int idx = p * 128 + tid;
      int r = idx >> 4, c8 = (idx & 15) * 8;
      const bf16_t* src = vhp + (size_t)(j0 + r) * D + c8;
      for (int e = 0; e < 8; ++e) Vs[(c8 + e) * 72 + r] = src[e];
    }
    wait_asynccnt0();
    __syncthreads();

    // ---- S = Q K^T  (16 rows x 64 cols, 16 WMMA) ----
    v8f sacc[4];
    for (int jt = 0; jt < 4; ++jt)
      for (int i = 0; i < 8; ++i) sacc[jt][i] = 0.f;
    for (int jt = 0; jt < 4; ++jt)
      for (int s = 0; s < 4; ++s) {
        v16bf bfr = load_frag_b(&Ks[(jt * 16) * 136 + s * 32], 136);
        sacc[jt] = wmma_bf16(qf[s], bfr, sacc[jt]);
      }

    // ---- scale (1/128), causal mask, running row-max ----
    float mnew[8];
    for (int i = 0; i < 8; ++i) mnew[i] = mrow[i];
    for (int jt = 0; jt < 4; ++jt)
      for (int i = 0; i < 8; ++i) {
        float sv = sacc[jt][i] * (1.0f / 128.0f);
        int gi = i0w + i + 8 * kh;
        int gj = j0 + jt * 16 + col;
        sv = (gj <= gi) ? sv : -1e30f;
        sacc[jt][i] = sv;
        mnew[i] = fmaxf(mnew[i], sv);
      }
    // reduce across the 16 lanes sharing each row (xor stays inside 16-group)
    for (int off = 1; off < 16; off <<= 1)
      for (int i = 0; i < 8; ++i)
        mnew[i] = fmaxf(mnew[i], __shfl_xor(mnew[i], off, 32));

    for (int i = 0; i < 8; ++i) {
      float alpha = __expf(mrow[i] - mnew[i]);
      mrow[i] = mnew[i];
      lrow[i] *= alpha;
      for (int dt = 0; dt < 8; ++dt) oacc[dt][i] *= alpha;
    }
    for (int jt = 0; jt < 4; ++jt)
      for (int i = 0; i < 8; ++i) {
        float pv = __expf(sacc[jt][i] - mrow[i]);
        lrow[i] += pv;
        sacc[jt][i] = pv;
      }

    // ---- P (D-layout) -> LDS -> A-fragment layout; O += P V ----
    bf16_t* pw = Ps[wave];
    for (int jt = 0; jt < 4; ++jt)
      for (int i = 0; i < 8; ++i)
        pw[(i + 8 * kh) * 72 + jt * 16 + col] = (bf16_t)sacc[jt][i];
    asm volatile("s_wait_dscnt 0" ::: "memory");  // cross-lane LDS RAW, same wave

    v16bf pf[2];
    for (int s = 0; s < 2; ++s) {
      Frag16 f;
      const bf16_t* pr = pw + col * 72 + s * 32;  // row = lane&15
      f.u[0] = *(const u32x4*)(pr + kh * 8);
      f.u[1] = *(const u32x4*)(pr + 16 + kh * 8);
      pf[s] = f.v;
    }
    for (int dt = 0; dt < 8; ++dt)
      for (int s = 0; s < 2; ++s) {
        v16bf vf = load_frag_b(&Vs[(dt * 16) * 72 + s * 32], 72);
        oacc[dt] = wmma_bf16(pf[s], vf, oacc[dt]);
      }
    __syncthreads();
  }

  // ---- finalize: reduce row sums across lanes, normalize, store ----
  for (int off = 1; off < 16; off <<= 1)
    for (int i = 0; i < 8; ++i) lrow[i] += __shfl_xor(lrow[i], off, 32);

  for (int dt = 0; dt < 8; ++dt)
    for (int i = 0; i < 8; ++i) {
      int t = i0w + i + 8 * kh;
      int d = dt * 16 + col;
      og[((size_t)b * T + t) * (H * D) + h * D + d] =
          (bf16_t)(oacc[dt][i] / lrow[i]);
    }
}

// ---------------------------------------------------------------------------
// Launch: QKV projections -> RoPE -> flash attention -> output projection.
// Workspace: q,k,v bf16 [B][H][T][D] + O bf16 [B][T][H*D]  (~67 MB)
// ---------------------------------------------------------------------------
extern "C" void kernel_launch(void* const* d_in, const int* in_sizes, int n_in,
                              void* d_out, int out_size, void* d_ws, size_t ws_size,
                              hipStream_t stream) {
  (void)in_sizes; (void)n_in; (void)out_size; (void)ws_size;
  const float* x  = (const float*)d_in[0];
  const float* wq = (const float*)d_in[1];
  const float* wk = (const float*)d_in[2];
  const float* wv = (const float*)d_in[3];
  const float* wo = (const float*)d_in[4];
  float* out = (float*)d_out;

  const size_t qkvElems = (size_t)2 * 16 * 2048 * 128;  // B*H*T*D
  bf16_t* qb = (bf16_t*)d_ws;
  bf16_t* kb = qb + qkvElems;
  bf16_t* vb = kb + qkvElems;
  bf16_t* ob = vb + qkvElems;  // [B][T][H*D]

  dim3 gg(16, 16, 2);  // (N/128, M/128, B)
  gemm2048_kernel<true, 0><<<gg, 256, 0, stream>>>((const void*)x, wq, (void*)qb);
  gemm2048_kernel<true, 0><<<gg, 256, 0, stream>>>((const void*)x, wk, (void*)kb);
  gemm2048_kernel<true, 0><<<gg, 256, 0, stream>>>((const void*)x, wv, (void*)vb);
  rope_kernel<<<dim3(2048, 16, 2), 64, 0, stream>>>(qb, kb);
  attn_kernel<<<dim3(32, 16, 2), 128, 0, stream>>>(qb, kb, vb, ob);
  gemm2048_kernel<false, 1><<<gg, 256, 0, stream>>>((const void*)ob, wo, (void*)out);
}
